// HGATLayer_6451040878649
// MI455X (gfx1250) — matrix operand order repaired and checked by reading, not compiled, not run on hardware
//
#include <hip/hip_runtime.h>
#include <hip/hip_bf16.h>
#include <math.h>

// ---------------------------------------------------------------------------
// Types for CDNA5 WMMA (wave32, 16x16x32 bf16 -> f32)
// ---------------------------------------------------------------------------
typedef __attribute__((ext_vector_type(16))) __bf16 v16bf;
typedef __attribute__((ext_vector_type(8)))  __bf16 v8bf;
typedef __attribute__((ext_vector_type(8)))  float  v8f;

#define DFEAT 256          // D = 256 (also H*C); GEMM K is always DFEAT
#define NHEAD 4
#define CCH   64
#define KSTEPS (DFEAT / 32)

// ---------------------------------------------------------------------------
// Helpers
// ---------------------------------------------------------------------------
__device__ __forceinline__ unsigned encodeOrd(float f) {
    unsigned u = __float_as_uint(f);
    return (u & 0x80000000u) ? ~u : (u | 0x80000000u);
}
__device__ __forceinline__ float decodeOrd(unsigned u) {
    u = (u & 0x80000000u) ? (u & 0x7FFFFFFFu) : ~u;
    return __uint_as_float(u);
}
#define ENC_NEGINF 0x007FFFFFu   // encodeOrd(-inf)

__device__ __forceinline__ float hash01(unsigned x) {
    x ^= x >> 16; x *= 0x7feb352dU;
    x ^= x >> 15; x *= 0x846ca68bU;
    x ^= x >> 16;
    return (float)(x >> 8) * (1.0f / 16777216.0f);
}

// ---------------------------------------------------------------------------
// Elementwise fp32 -> bf16 cast
// ---------------------------------------------------------------------------
__global__ __launch_bounds__(256) void cast_bf16_kernel(
    const float* __restrict__ in, __bf16* __restrict__ out, size_t n)
{
    size_t i = (size_t)blockIdx.x * blockDim.x + threadIdx.x;
    if (i < n) out[i] = (__bf16)in[i];
}

// W [K x N] fp32 (row-major) -> Wt [N x K] bf16 (row-major)
__global__ __launch_bounds__(256) void transpose_cast_kernel(
    const float* __restrict__ in, __bf16* __restrict__ out, int K, int N)
{
    int i = blockIdx.x * blockDim.x + threadIdx.x;
    if (i >= K * N) return;
    int k = i / N, n = i % N;
    out[(size_t)n * K + k] = (__bf16)in[i];
}

__global__ __launch_bounds__(256) void fill_u32_kernel(
    unsigned* __restrict__ p, unsigned v, size_t n)
{
    size_t i = (size_t)blockIdx.x * blockDim.x + threadIdx.x;
    if (i < n) p[i] = v;
}

// ---------------------------------------------------------------------------
// Fragment loaders (CDNA5 bf16 WMMA register layouts)
// ---------------------------------------------------------------------------
// A 16x32: lane group hi: VGPR0-3 K = hi*8..+7, VGPR4-7 K = 16+hi*8..+7
__device__ __forceinline__ v16bf load_afrag(const __bf16* __restrict__ arow,
                                            int k0, int hi)
{
    v8bf lo = *(const v8bf*)(arow + k0 + hi * 8);
    v8bf hiv = *(const v8bf*)(arow + k0 + 16 + hi * 8);
    v16bf a;
#pragma unroll
    for (int i = 0; i < 8; ++i) { a[i] = lo[i]; a[8 + i] = hiv[i]; }
    return a;
}
// B 32x16 (from pre-transposed [N][K]): lane group hi: K = hi*8..+7, 16+hi*8..+7
__device__ __forceinline__ v16bf load_bfrag(const __bf16* __restrict__ brow,
                                            int k0, int hi)
{
    v8bf b0 = *(const v8bf*)(brow + k0 + hi * 8);
    v8bf b1 = *(const v8bf*)(brow + k0 + 16 + hi * 8);
    v16bf b;
#pragma unroll
    for (int i = 0; i < 8; ++i) { b[i] = b0[i]; b[8 + i] = b1[i]; }
    return b;
}

// ---------------------------------------------------------------------------
// WMMA GEMM: C[M,N] = A[M,256](bf16) @ Bt[N,256]^T(bf16) + bias[N], f32 out.
// One wave -> 16x64 tile. K fully unrolled (8 steps), one-step prefetch with
// sched_barrier pinning the load group ahead of the previous step's WMMAs so
// the scheduler cannot sink the prefetch and collapse the pipeline.
// ---------------------------------------------------------------------------
__global__ __launch_bounds__(128) void wmma_gemm_bias_kernel(
    const __bf16* __restrict__ A, const __bf16* __restrict__ Bt,
    const float* __restrict__ bias, float* __restrict__ C,
    int M, int N)
{
    const int lane = threadIdx.x & 31;
    const int gw   = blockIdx.x * 4 + (threadIdx.x >> 5);
    const int ntl  = N >> 6;                 // 64-wide n-tiles
    const int tm   = gw / ntl;
    const int tn   = gw % ntl;
    if (tm * 16 >= M) return;

    const int mrow = lane & 15;
    const int hi   = lane >> 4;

    const __bf16* arow = A + (size_t)(tm * 16 + mrow) * DFEAT;
    const __bf16* brow[4];
#pragma unroll
    for (int s = 0; s < 4; ++s)
        brow[s] = Bt + (size_t)(tn * 64 + s * 16 + mrow) * DFEAT;

    v8f acc[4] = {};

    v16bf afr[KSTEPS];
    v16bf bfr[KSTEPS][4];
    // prologue: fragments for k-step 0
    afr[0] = load_afrag(arow, 0, hi);
#pragma unroll
    for (int s = 0; s < 4; ++s) bfr[0][s] = load_bfrag(brow[s], 0, hi);

#pragma unroll
    for (int ks = 0; ks < KSTEPS; ++ks) {
        if (ks + 1 < KSTEPS) {
            const int k1 = (ks + 1) << 5;
            afr[ks + 1] = load_afrag(arow, k1, hi);
#pragma unroll
            for (int s = 0; s < 4; ++s)
                bfr[ks + 1][s] = load_bfrag(brow[s], k1, hi);
        }
        // Pin: prefetch loads above must issue before the WMMAs below.
        __builtin_amdgcn_sched_barrier(0);
#pragma unroll
        for (int s = 0; s < 4; ++s)
            acc[s] = __builtin_amdgcn_wmma_f32_16x16x32_bf16(
                false, afr[ks], false, bfr[ks][s], (short)0, acc[s],
                false, false);
        __builtin_amdgcn_sched_barrier(0);
    }

#pragma unroll
    for (int s = 0; s < 4; ++s) {
        int ncol = tn * 64 + s * 16 + mrow;          // n = lane&15
        float bv = bias ? bias[ncol] : 0.0f;
#pragma unroll
        for (int v = 0; v < 8; ++v) {
            int r = tm * 16 + v + 8 * hi;            // C/D VGPR layout
            C[(size_t)r * N + ncol] = acc[s][v] + bv;
        }
    }
}

// ---------------------------------------------------------------------------
// Edge pass 1: per-edge scores e[h] = sum_c lrelu(xl[s]+xr[d]) * att, and
// segment max via ordered-uint atomicMax. One wave per edge, 8 lanes/head.
// ---------------------------------------------------------------------------
__global__ __launch_bounds__(256) void edge_scores_kernel(
    const float* __restrict__ xl, const float* __restrict__ xr,
    const int* __restrict__ src, const int* __restrict__ dst,
    const float* __restrict__ att,          // [4*64] flat
    float* __restrict__ esc,                // [nE*4]
    unsigned* __restrict__ menc,            // [n_dst*4]
    int nE, int nReal)
{
    int wid  = (int)(((size_t)blockIdx.x * blockDim.x + threadIdx.x) >> 5);
    int lane = threadIdx.x & 31;
    if (wid >= nE) return;
    int s, d;
    if (wid < nReal) { s = src[wid]; d = dst[wid]; }
    else             { s = d = wid - nReal; }        // self-loop
    const float* pl = xl + (size_t)s * DFEAT;
    const float* pr = xr + (size_t)d * DFEAT;
    int h    = lane >> 3;
    int base = lane * 8;
    float acc = 0.0f;
#pragma unroll
    for (int i = 0; i < 8; ++i) {
        int idx = base + i;
        float v = pl[idx] + pr[idx];
        v = v > 0.0f ? v : 0.2f * v;                 // leaky_relu(0.2)
        acc += v * att[idx];
    }
    acc += __shfl_xor(acc, 1, 32);
    acc += __shfl_xor(acc, 2, 32);
    acc += __shfl_xor(acc, 4, 32);
    if ((lane & 7) == 0) {
        esc[(size_t)wid * 4 + h] = acc;
        atomicMax(&menc[(size_t)d * 4 + h], encodeOrd(acc));
    }
}

// Decode segment max in place (non-finite / untouched -> 0, like reference)
__global__ __launch_bounds__(256) void segmax_finalize_kernel(
    unsigned* __restrict__ menc, size_t n)
{
    size_t i = (size_t)blockIdx.x * blockDim.x + threadIdx.x;
    if (i >= n) return;
    unsigned e = menc[i];
    float m = (e <= ENC_NEGINF) ? 0.0f : decodeOrd(e);
    if (!isfinite(m)) m = 0.0f;
    ((float*)menc)[i] = m;
}

// Edge pass 2: ex = exp(e - m[dst]); accumulate denominator
__global__ __launch_bounds__(256) void edge_exp_kernel(
    float* __restrict__ esc, const float* __restrict__ m,
    float* __restrict__ den,
    const int* __restrict__ src, const int* __restrict__ dst,
    int nE, int nReal)
{
    size_t t = (size_t)blockIdx.x * blockDim.x + threadIdx.x;
    int e = (int)(t >> 2);
    int h = (int)(t & 3);
    if (e >= nE) return;
    int d = (e < nReal) ? dst[e] : (e - nReal);
    float ex = expf(esc[(size_t)e * 4 + h] - m[(size_t)d * 4 + h]);
    esc[(size_t)e * 4 + h] = ex;
    atomicAdd(&den[(size_t)d * 4 + h], ex);
}

// Edge pass 3: out[dst] += alpha * xl[src], scatter with f32 atomics
__global__ __launch_bounds__(256) void edge_aggregate_kernel(
    const float* __restrict__ xl, const float* __restrict__ esc,
    const float* __restrict__ den,
    const int* __restrict__ src, const int* __restrict__ dst,
    float* __restrict__ agg,                 // [n_dst * 256]
    int nE, int nReal)
{
    int wid  = (int)(((size_t)blockIdx.x * blockDim.x + threadIdx.x) >> 5);
    int lane = threadIdx.x & 31;
    if (wid >= nE) return;
    int s, d;
    if (wid < nReal) { s = src[wid]; d = dst[wid]; }
    else             { s = d = wid - nReal; }
    int h = lane >> 3;
    float alpha = esc[(size_t)wid * 4 + h] /
                  (den[(size_t)d * 4 + h] + 1e-16f);
    const float* pl = xl + (size_t)s * DFEAT;
    float* pa = agg + (size_t)d * DFEAT;
    int base = lane * 8;
#pragma unroll
    for (int i = 0; i < 8; ++i)
        atomicAdd(&pa[base + i], alpha * pl[base + i]);
}

// ---------------------------------------------------------------------------
// Fused tail: head-mean + bias (+relation mean scale) -> ELU -> +proj ->
// LayerNorm(64) -> MC dropout. One wave per node row (2 channels/lane).
// ---------------------------------------------------------------------------
__global__ __launch_bounds__(256) void final_node_kernel(
    const float* __restrict__ agg,   // [n * 4 * 64]
    const float* __restrict__ b1, const float* __restrict__ b2,
    float scale,
    const float* __restrict__ proj,  // [n * 64]
    const float* __restrict__ lng, const float* __restrict__ lnb,
    float* __restrict__ out, int n, unsigned seed)
{
    int row  = (int)(((size_t)blockIdx.x * blockDim.x + threadIdx.x) >> 5);
    int lane = threadIdx.x & 31;
    if (row >= n) return;
    const float* pa = agg + (size_t)row * DFEAT;
    float v[2];
#pragma unroll
    for (int j = 0; j < 2; ++j) {
        int c = lane + j * 32;
        float s4 = pa[c] + pa[64 + c] + pa[128 + c] + pa[192 + c];
        float pre = scale * (0.25f * s4 + b1[c] + (b2 ? b2[c] : 0.0f));
        float e = pre > 0.0f ? pre : (expf(pre) - 1.0f);   // ELU
        v[j] = e + proj[(size_t)row * CCH + c];
    }
    float sum = v[0] + v[1];
#pragma unroll
    for (int o = 16; o >= 1; o >>= 1) sum += __shfl_xor(sum, o, 32);
    float mu = sum * (1.0f / 64.0f);
    float d0 = v[0] - mu, d1 = v[1] - mu;
    float sq = d0 * d0 + d1 * d1;
#pragma unroll
    for (int o = 16; o >= 1; o >>= 1) sq += __shfl_xor(sq, o, 32);
    float rs = rsqrtf(sq * (1.0f / 64.0f) + 1e-5f);
#pragma unroll
    for (int j = 0; j < 2; ++j) {
        int c = lane + j * 32;
        float y = (v[j] - mu) * rs * lng[c] + lnb[c];
        unsigned idx = (unsigned)((size_t)row * CCH + c);
        float r = hash01(seed ^ (idx * 2654435761u));
        y = (r < 0.8f) ? y * 1.25f : 0.0f;   // keep prob 0.8, scale 1/0.8
        out[(size_t)row * CCH + c] = y;
    }
}

// ---------------------------------------------------------------------------
// Host launcher
// ---------------------------------------------------------------------------
extern "C" void kernel_launch(void* const* d_in, const int* in_sizes, int n_in,
                              void* d_out, int out_size, void* d_ws, size_t ws_size,
                              hipStream_t stream)
{
    const float* x_met = (const float*)d_in[0];
    const float* x_rxn = (const float*)d_in[1];
    const int* sub_src = (const int*)d_in[2];
    const int* sub_dst = (const int*)d_in[3];
    const int* prod_src = (const int*)d_in[4];
    const int* prod_dst = (const int*)d_in[5];
    const int* shr_src = (const int*)d_in[6];
    const int* shr_dst = (const int*)d_in[7];
    // relation params: sub(8..13), prod(14..19), shr(20..25)
    const float* Wl_sub = (const float*)d_in[8];
    const float* bl_sub = (const float*)d_in[9];
    const float* Wr_sub = (const float*)d_in[10];
    const float* br_sub = (const float*)d_in[11];
    const float* att_sub = (const float*)d_in[12];
    const float* b_sub = (const float*)d_in[13];
    const float* Wl_prod = (const float*)d_in[14];
    const float* bl_prod = (const float*)d_in[15];
    const float* Wr_prod = (const float*)d_in[16];
    const float* br_prod = (const float*)d_in[17];
    const float* att_prod = (const float*)d_in[18];
    const float* b_prod = (const float*)d_in[19];
    const float* Wl_shr = (const float*)d_in[20];
    const float* bl_shr = (const float*)d_in[21];
    const float* Wr_shr = (const float*)d_in[22];
    const float* br_shr = (const float*)d_in[23];
    const float* att_shr = (const float*)d_in[24];
    const float* b_shr = (const float*)d_in[25];
    const float* W_proj_rxn = (const float*)d_in[26];
    const float* b_proj_rxn = (const float*)d_in[27];
    const float* W_proj_met = (const float*)d_in[28];
    const float* b_proj_met = (const float*)d_in[29];
    const float* ln_rxn_g = (const float*)d_in[30];
    const float* ln_rxn_b = (const float*)d_in[31];
    const float* ln_met_g = (const float*)d_in[32];
    const float* ln_met_b = (const float*)d_in[33];

    const int E = in_sizes[2];
    const int NMET = in_sizes[0] / DFEAT;
    const int NRXN = in_sizes[1] / DFEAT;

    // workspace layout
    char* ws = (char*)d_ws;
    size_t off = 0;
    auto walloc = [&](size_t bytes) -> void* {
        void* p = ws + off;
        off = (off + bytes + 255) & ~(size_t)255;
        return p;
    };
    __bf16* xmet_bf = (__bf16*)walloc((size_t)NMET * DFEAT * 2);
    __bf16* xrxn_bf = (__bf16*)walloc((size_t)NRXN * DFEAT * 2);
    __bf16* wt      = (__bf16*)walloc((size_t)DFEAT * DFEAT * 2);
    float*  xl      = (float*)walloc((size_t)NRXN * DFEAT * 4);
    float*  xr      = (float*)walloc((size_t)NRXN * DFEAT * 4);
    float*  esc     = (float*)walloc((size_t)(E + NRXN) * NHEAD * 4);
    unsigned* menc  = (unsigned*)walloc((size_t)NRXN * NHEAD * 4);
    float*  den     = (float*)walloc((size_t)NRXN * NHEAD * 4);
    float*  agg_rxn = (float*)walloc((size_t)NRXN * DFEAT * 4);
    float*  agg_met = (float*)walloc((size_t)NMET * DFEAT * 4);
    float*  proj_r  = (float*)walloc((size_t)NRXN * CCH * 4);
    float*  proj_m  = (float*)walloc((size_t)NMET * CCH * 4);

    auto cdiv = [](size_t a, size_t b) { return (unsigned)((a + b - 1) / b); };

    // bf16 activations
    cast_bf16_kernel<<<cdiv((size_t)NMET * DFEAT, 256), 256, 0, stream>>>(
        x_met, xmet_bf, (size_t)NMET * DFEAT);
    cast_bf16_kernel<<<cdiv((size_t)NRXN * DFEAT, 256), 256, 0, stream>>>(
        x_rxn, xrxn_bf, (size_t)NRXN * DFEAT);

    auto run_gemm = [&](const __bf16* A, const float* W, const float* b,
                        float* Cout, int M, int N) {
        transpose_cast_kernel<<<cdiv((size_t)DFEAT * N, 256), 256, 0, stream>>>(
            W, wt, DFEAT, N);
        int waves = (M / 16) * (N / 64);
        wmma_gemm_bias_kernel<<<cdiv((size_t)waves, 4), 128, 0, stream>>>(
            A, wt, b, Cout, M, N);
    };

    auto run_relation = [&](const __bf16* Asrc, int Msrc,
                            const __bf16* Adst, int Mdst,
                            const float* Wl, const float* bl,
                            const float* Wr, const float* br,
                            const float* att,
                            const int* srcI, const int* dstI,
                            int nE, int nReal, float* agg) {
        run_gemm(Asrc, Wl, bl, xl, Msrc, DFEAT);
        run_gemm(Adst, Wr, br, xr, Mdst, DFEAT);
        size_t nm = (size_t)Mdst * NHEAD;
        fill_u32_kernel<<<cdiv(nm, 256), 256, 0, stream>>>(menc, ENC_NEGINF, nm);
        fill_u32_kernel<<<cdiv(nm, 256), 256, 0, stream>>>((unsigned*)den, 0u, nm);
        edge_scores_kernel<<<cdiv((size_t)nE, 8), 256, 0, stream>>>(
            xl, xr, srcI, dstI, att, esc, menc, nE, nReal);
        segmax_finalize_kernel<<<cdiv(nm, 256), 256, 0, stream>>>(menc, nm);
        edge_exp_kernel<<<cdiv((size_t)nE * 4, 256), 256, 0, stream>>>(
            esc, (const float*)menc, den, srcI, dstI, nE, nReal);
        edge_aggregate_kernel<<<cdiv((size_t)nE, 8), 256, 0, stream>>>(
            xl, esc, den, srcI, dstI, agg, nE, nReal);
    };

    // zero accumulation buffers
    fill_u32_kernel<<<cdiv((size_t)NRXN * DFEAT, 256), 256, 0, stream>>>(
        (unsigned*)agg_rxn, 0u, (size_t)NRXN * DFEAT);
    fill_u32_kernel<<<cdiv((size_t)NMET * DFEAT, 256), 256, 0, stream>>>(
        (unsigned*)agg_met, 0u, (size_t)NMET * DFEAT);

    // substrate: met -> rxn
    run_relation(xmet_bf, NMET, xrxn_bf, NRXN,
                 Wl_sub, bl_sub, Wr_sub, br_sub, att_sub,
                 sub_src, sub_dst, E, E, agg_rxn);
    // shared_metabolite: rxn -> rxn, with self loops appended
    run_relation(xrxn_bf, NRXN, xrxn_bf, NRXN,
                 Wl_shr, bl_shr, Wr_shr, br_shr, att_shr,
                 shr_src, shr_dst, E + NRXN, E, agg_rxn);
    // product: rxn -> met
    run_relation(xrxn_bf, NRXN, xmet_bf, NMET,
                 Wl_prod, bl_prod, Wr_prod, br_prod, att_prod,
                 prod_src, prod_dst, E, E, agg_met);

    // residual projections
    run_gemm(xrxn_bf, W_proj_rxn, b_proj_rxn, proj_r, NRXN, CCH);
    run_gemm(xmet_bf, W_proj_met, b_proj_met, proj_m, NMET, CCH);

    // fused tail -> d_out (rxn first, then met)
    float* out_rxn = (float*)d_out;
    float* out_met = out_rxn + (size_t)NRXN * CCH;
    final_node_kernel<<<cdiv((size_t)NRXN, 8), 256, 0, stream>>>(
        agg_rxn, b_sub, b_shr, 0.5f, proj_r, ln_rxn_g, ln_rxn_b,
        out_rxn, NRXN, 0x9E3779B9u);
    final_node_kernel<<<cdiv((size_t)NMET, 8), 256, 0, stream>>>(
        agg_met, b_prod, nullptr, 1.0f, proj_m, ln_met_g, ln_met_b,
        out_met, NMET, 0x85EBCA6Bu);
}
